// TitansMACBlock_9079560864299
// MI455X (gfx1250) — compile-verified, matrix operand order checked by reading, not run on hardware
//
#include <hip/hip_runtime.h>
#include <hip/hip_bf16.h>

// ---------------- problem constants (match reference) ----------------
#define D_   1024
#define H_   16
#define DH_  64
#define T_   1024
#define B_   2
#define P_   4
#define WIN_ 256
#define MH_  4096
#define FF_  4096
#define L_   (P_ + 2*T_)        // 2052
#define ROWS_ (B_*L_)           // 4104
#define QKV_PAD_ 64             // zeroed padding rows after qkv (clamp-free attention)

// ---------------- WMMA types ----------------
typedef __attribute__((ext_vector_type(16))) __bf16 v16bf;
typedef __attribute__((ext_vector_type(8)))  float  v8f;

union frag16 { v16bf v; uint4 q[2]; unsigned short s[16]; };

__device__ inline v8f wmma_bf16(v16bf a, v16bf b, v8f c) {
  // D = A(16x32 bf16) * B(32x16 bf16) + C(16x16 f32)
  return __builtin_amdgcn_wmma_f32_16x16x32_bf16(false, a, false, b, (short)0, c,
                                                 false, false);
}

__device__ inline unsigned short f2bf(float f) {
  unsigned u = __float_as_uint(f);
  u += 0x7FFFu + ((u >> 16) & 1u);           // round-to-nearest-even
  return (unsigned short)(u >> 16);
}
__device__ inline int imin(int a, int b) { return a < b ? a : b; }
__device__ inline int imax(int a, int b) { return a > b ? a : b; }

// ---- CDNA5 async global->LDS copy (ASYNCcnt-tracked, no VGPR round trip) ----
// Generic pointers to __shared__ carry the LDS offset in their low 32 bits
// (flat LDS aperture), so this is a valid VDST address.
__device__ inline void async_lds_load_b128(unsigned lds_off, const void* gptr) {
  unsigned long long ga = (unsigned long long)(size_t)gptr;
  asm volatile("global_load_async_to_lds_b128 %0, %1, off"
               :: "v"(lds_off), "v"(ga) : "memory");
}
__device__ inline void wait_async0() {
  asm volatile("s_wait_asynccnt 0x0" ::: "memory");
}

// ---------------- f32 -> bf16 conversion ----------------
__global__ __launch_bounds__(256) void cvt_bf16_kernel(
    const float* __restrict__ in, unsigned short* __restrict__ out, long n) {
  long i = ((long)blockIdx.x * 256 + threadIdx.x) * 4;
  if (i + 3 < n) {
    float4 v = *(const float4*)(in + i);
    out[i + 0] = f2bf(v.x); out[i + 1] = f2bf(v.y);
    out[i + 2] = f2bf(v.z); out[i + 3] = f2bf(v.w);
  } else {
    for (long k = i; k < n; ++k) out[k] = f2bf(in[k]);
  }
}

// ---------------- zero a bf16 region (qkv padding rows) ----------------
__global__ __launch_bounds__(256) void zero_bf16_kernel(
    unsigned short* __restrict__ p, long n) {
  long i = ((long)blockIdx.x * 256 + threadIdx.x) * 8;
  if (i + 7 < n) *(uint4*)(p + i) = uint4{0, 0, 0, 0};
}

// ---------------- build aug = concat(pmem, mem_ctx, x) (f32) ----------------
__global__ __launch_bounds__(256) void build_aug_kernel(
    const float* __restrict__ x, const float* __restrict__ pmem,
    const float* __restrict__ mctx, float* __restrict__ aug) {
  int r = blockIdx.x;                    // 0 .. B*L-1
  int b = r / L_, l = r % L_;
  const float* src;
  if (l < P_)            src = pmem + (long)l * D_;
  else if (l < P_ + T_)  src = mctx + ((long)b * T_ + (l - P_)) * D_;
  else                   src = x    + ((long)b * T_ + (l - P_ - T_)) * D_;
  float* dst = aug + (long)r * D_;
  for (int i = threadIdx.x; i < D_; i += 256) dst[i] = src[i];
}

// ---------------- LayerNorm (f32 in -> bf16 out) ----------------
__global__ __launch_bounds__(256) void ln_kernel(
    const float* __restrict__ in, const float* __restrict__ g,
    const float* __restrict__ bt, unsigned short* __restrict__ out) {
  __shared__ float red[256];
  int row = blockIdx.x, tid = threadIdx.x;
  const float* x = in + (long)row * D_;
  float loc[4]; float s = 0.f;
#pragma unroll
  for (int i = 0; i < 4; ++i) { loc[i] = x[tid + i * 256]; s += loc[i]; }
  red[tid] = s; __syncthreads();
  for (int off = 128; off > 0; off >>= 1) {
    if (tid < off) red[tid] += red[tid + off];
    __syncthreads();
  }
  float mean = red[0] * (1.0f / D_);
  __syncthreads();
  float v = 0.f;
#pragma unroll
  for (int i = 0; i < 4; ++i) { float d = loc[i] - mean; v += d * d; }
  red[tid] = v; __syncthreads();
  for (int off = 128; off > 0; off >>= 1) {
    if (tid < off) red[tid] += red[tid + off];
    __syncthreads();
  }
  float rs = rsqrtf(red[0] * (1.0f / D_) + 1e-5f);
#pragma unroll
  for (int i = 0; i < 4; ++i) {
    int c = tid + i * 256;
    out[(long)row * D_ + c] = f2bf((loc[i] - mean) * rs * g[c] + bt[c]);
  }
}

// ---------------- LDS-tiled bf16 WMMA GEMM:  out = epi(A[M,K] * W[N,K]^T + bias) -----
// Weight tile staged via async global->LDS DMA, explicit double buffer: each
// half-step waits its buffer, launches the next tile's DMA into the other
// buffer, then runs its 8 WMMAs (DMA s+1 overlaps compute s).
enum { EPI_RELU_BF16 = 0, EPI_BIAS_F32 = 1, EPI_BIAS_BF16 = 2,
       EPI_RES_F32 = 3, EPI_GELU_BF16 = 4 };

#define LWT 2560   // halves per weight buffer: 64 rows x (32+8 pad)

template <int EPI>
__global__ __launch_bounds__(256) void gemm_kernel(
    const unsigned short* __restrict__ A, const unsigned short* __restrict__ W,
    const float* __restrict__ bias, const float* __restrict__ res,
    void* __restrict__ out, int M, int N, int K) {
  __shared__ __align__(16) unsigned short lw[2 * LWT];
  int tid = threadIdx.x;
  int lane = tid & 31;
  int wave = tid >> 5;                     // 0..7 -> M sub-tile
  int col  = lane & 15;
  int kl   = (lane >> 4) * 8;              // K offset of lane half (also C-row offset)
  int m0 = blockIdx.x * 256 + wave * 32;
  int n0 = blockIdx.y * 64;

  v8f acc[2][4] = {};
  int ar[2];
  ar[0] = imin(m0 +      col, M - 1);
  ar[1] = imin(m0 + 16 + col, M - 1);

  int wn = tid >> 2;                       // 0..63 : weight row within tile
  int wc = (tid & 3) * 8;                  // 0,8,16,24 : K chunk (halves)
  const unsigned short* wrow = W + (long)(n0 + wn) * K + wc;
  unsigned ldsoff[2];
  ldsoff[0] = (unsigned)(size_t)&lw[wn * 40 + wc];
  ldsoff[1] = ldsoff[0] + LWT * 2;

  auto kstep = [&](int kb, const unsigned short* lb) {
    frag16 Af[2];
#pragma unroll
    for (int mf = 0; mf < 2; ++mf) {
      const unsigned short* ap = A + (size_t)(unsigned)(ar[mf] * K + kb + kl);
      Af[mf].q[0] = *(const uint4*)ap;
      Af[mf].q[1] = *(const uint4*)(ap + 16);
    }
    if (kb + 32 < K)
      __builtin_prefetch(A + (size_t)(unsigned)(ar[0] * K + kb + 32 + kl), 0, 1);
#pragma unroll
    for (int nf = 0; nf < 4; ++nf) {
      frag16 Bf;
      int nl = nf * 16 + col;
      Bf.q[0] = *(const uint4*)&lb[nl * 40 + kl];
      Bf.q[1] = *(const uint4*)&lb[nl * 40 + kl + 16];
#pragma unroll
      for (int mf = 0; mf < 2; ++mf)
        acc[mf][nf] = wmma_bf16(Af[mf].v, Bf.v, acc[mf][nf]);
    }
  };

  int nst = K >> 5;                        // K/32: 32 or 128 (always even)
  async_lds_load_b128(ldsoff[0], wrow);
  for (int s = 0; s < nst; s += 2) {
    wait_async0();
    __syncthreads();
    if (s + 1 < nst) async_lds_load_b128(ldsoff[1], wrow + (s + 1) * 32);
    kstep(s * 32, lw);
    wait_async0();
    __syncthreads();
    if (s + 2 < nst) async_lds_load_b128(ldsoff[0], wrow + (s + 2) * 32);
    kstep((s + 1) * 32, lw + LWT);
  }

  // epilogue
#pragma unroll
  for (int nf = 0; nf < 4; ++nf) {
    int n = n0 + nf * 16 + col;
    float bv = bias[n];
#pragma unroll
    for (int mf = 0; mf < 2; ++mf) {
      int mbase = m0 + mf * 16 + kl;       // C layout: row = mbase + i
#pragma unroll
      for (int i = 0; i < 8; ++i) {
        int mm = mbase + i;
        if (mm >= M) continue;
        float v = acc[mf][nf][i] + bv;
        long off = (long)mm * N + n;
        if (EPI == EPI_RELU_BF16)
          ((unsigned short*)out)[off] = f2bf(fmaxf(v, 0.f));
        else if (EPI == EPI_BIAS_F32)
          ((float*)out)[off] = v;
        else if (EPI == EPI_BIAS_BF16)
          ((unsigned short*)out)[off] = f2bf(v);
        else if (EPI == EPI_RES_F32)
          ((float*)out)[off] = v + res[off];
        else if (EPI == EPI_GELU_BF16)
          ((unsigned short*)out)[off] =
              f2bf(0.5f * v * (1.f + erff(v * 0.70710678f)));
      }
    }
  }
}

// ---------------- sliding-window attention (flash-style, one wave / 16 queries) -----
// qkv: bf16 (B*L + pad, 3D) rows, pad rows zeroed -> no row clamping needed.
// ctx out: bf16 (B*L, D)
__global__ __launch_bounds__(32) void attn_kernel(
    const unsigned short* __restrict__ qkv, unsigned short* __restrict__ ctx) {
  __shared__ __align__(16) unsigned short vt[DH_][40]; // V^T tile: 64 dims x 32 keys
  int lane = threadIdx.x;
  int qt = blockIdx.x, h = blockIdx.y, b = blockIdx.z;
  int q0 = qt * 16;
  int col = lane & 15;                     // query (or d-row) within 16
  int kl  = (lane >> 4) * 8;               // K-dim offset of lane half
  const int rs3 = 3 * D_;                  // qkv row stride (halves)
  const unsigned short* qbase = qkv + (long)b * L_ * rs3 + h * DH_;
  const unsigned short* kbase = qbase + D_;
  const unsigned short* vbase = qbase + 2 * D_;

  int qg = q0 + col;                       // logical query index (may exceed L-1)

  // Q fragments (B layout for S^T = K * Q^T), d-chunks of 32
  frag16 Qf[2];
#pragma unroll
  for (int c = 0; c < 2; ++c) {
    const unsigned short* qp = qbase + (size_t)(unsigned)(qg * rs3 + c * 32 + kl);
    Qf[c].q[0] = *(const uint4*)qp;
    Qf[c].q[1] = *(const uint4*)(qp + 16);
  }

  float m = -1e30f, lsum = 0.f;
  v8f O[4] = {};                           // O^T accumulators: 4 d-tiles x (16x16)

  int vkey = (lane >> 3);                  // V stage: base key lane handles
  int vch  = lane & 7;                     // 8-half chunk of the V row

  int kb0 = imax(0, q0 - (WIN_ - 1)) & ~31;
  int kend = q0 + 15;
  for (int kb = kb0; kb <= kend; kb += 32) {
    // stage V^T (32 keys x 64 dims) transposed into LDS
    __syncthreads();
#pragma unroll
    for (int it = 0; it < 8; ++it) {
      int key = it * 4 + vkey;             // 0..31
      union { uint4 q; unsigned short s[8]; } u;
      u.q = *(const uint4*)(vbase + (size_t)(unsigned)((kb + key) * rs3 + vch * 8));
#pragma unroll
      for (int e = 0; e < 8; ++e) vt[vch * 8 + e][key] = u.s[e];
    }
    __syncthreads();

    // wave-uniform: is this 32-key tile fully inside every query's window?
    bool full = (kb + 31 <= q0) && (kb > q0 + 15 - WIN_);

    // two score fragments S^T (16 keys x 16 queries each)
    float p0[8], p1[8];
    float tmax = -1e30f;
#pragma unroll
    for (int sfr = 0; sfr < 2; ++sfr) {
      int ks = kb + sfr * 16;
      v8f sacc = {};
#pragma unroll
      for (int c = 0; c < 2; ++c) {
        frag16 Kf;
        const unsigned short* kp =
            kbase + (size_t)(unsigned)((ks + col) * rs3 + c * 32 + kl);
        Kf.q[0] = *(const uint4*)kp;
        Kf.q[1] = *(const uint4*)(kp + 16);
        sacc = wmma_bf16(Kf.v, Qf[c].v, sacc);
      }
      float* pp = sfr ? p1 : p0;
      if (full) {
#pragma unroll
        for (int i = 0; i < 8; ++i) {
          float sv = sacc[i] * 0.125f;     // 1/sqrt(64)
          pp[i] = sv;
          tmax = fmaxf(tmax, sv);
        }
      } else {
#pragma unroll
        for (int i = 0; i < 8; ++i) {
          int j = ks + i + kl;             // key row of this C element
          float sv = sacc[i] * 0.125f;
          bool ok = (j <= qg) && (j > qg - WIN_);
          sv = ok ? sv : -1e30f;
          pp[i] = sv;
          tmax = fmaxf(tmax, sv);
        }
      }
    }
    tmax = fmaxf(tmax, __shfl_xor(tmax, 16, 32));
    float mnew = fmaxf(m, tmax);
    float corr = __expf(m - mnew);
    float tsum = 0.f;
    if (full) {
#pragma unroll
      for (int i = 0; i < 8; ++i) {
        p0[i] = __expf(p0[i] - mnew); tsum += p0[i];
        p1[i] = __expf(p1[i] - mnew); tsum += p1[i];
      }
    } else {
#pragma unroll
      for (int i = 0; i < 8; ++i) {
        p0[i] = (p0[i] <= -1e29f) ? 0.f : __expf(p0[i] - mnew); tsum += p0[i];
        p1[i] = (p1[i] <= -1e29f) ? 0.f : __expf(p1[i] - mnew); tsum += p1[i];
      }
    }
    tsum += __shfl_xor(tsum, 16, 32);
    lsum = lsum * corr + tsum;
    m = mnew;
#pragma unroll
    for (int f = 0; f < 4; ++f)
#pragma unroll
      for (int i = 0; i < 8; ++i) O[f][i] *= corr;

    // pack P^T into B fragment (lane-local thanks to S^T layout)
    frag16 Bp;
#pragma unroll
    for (int i = 0; i < 8; ++i) { Bp.s[i] = f2bf(p0[i]); Bp.s[8 + i] = f2bf(p1[i]); }

    // O^T += V^T * P^T   (4 d-tiles)
#pragma unroll
    for (int f = 0; f < 4; ++f) {
      frag16 Vf;
      int drow = f * 16 + col;
      Vf.q[0] = *(const uint4*)&vt[drow][kl];
      Vf.q[1] = *(const uint4*)&vt[drow][kl + 16];
      O[f] = wmma_bf16(Vf.v, Bp.v, O[f]);
    }
  }

  if (qg < L_) {
    float inv = 1.0f / lsum;
    unsigned short* crow = ctx + ((long)b * L_ + qg) * D_ + h * DH_;
#pragma unroll
    for (int f = 0; f < 4; ++f)
#pragma unroll
      for (int i = 0; i < 8; ++i) {
        int d = f * 16 + kl + i;
        crow[d] = f2bf(O[f][i] * inv);
      }
  }
}

// ---------------- launch sequence ----------------
extern "C" void kernel_launch(void* const* d_in, const int* in_sizes, int n_in,
                              void* d_out, int out_size, void* d_ws, size_t ws_size,
                              hipStream_t stream) {
  const float* x    = (const float*)d_in[0];
  const float* pmem = (const float*)d_in[1];
  const float* mW1  = (const float*)d_in[2];
  const float* mb1  = (const float*)d_in[3];
  const float* mW2  = (const float*)d_in[4];
  const float* mb2  = (const float*)d_in[5];
  const float* ln1g = (const float*)d_in[6];
  const float* ln1b = (const float*)d_in[7];
  const float* Wqkv = (const float*)d_in[8];
  const float* bqkv = (const float*)d_in[9];
  const float* Wo   = (const float*)d_in[10];
  const float* bo   = (const float*)d_in[11];
  const float* ln2g = (const float*)d_in[12];
  const float* ln2b = (const float*)d_in[13];
  const float* fW1  = (const float*)d_in[14];
  const float* fb1  = (const float*)d_in[15];
  const float* fW2  = (const float*)d_in[16];
  const float* fb2  = (const float*)d_in[17];
  float* out = (float*)d_out;

  char* ws = (char*)d_ws;
  size_t off = 0;
  auto alloc = [&](size_t bytes) -> void* {
    void* p = ws + off;
    off = (off + bytes + 255) & ~(size_t)255;
    return p;
  };
  typedef unsigned short bf;
  bf* xb     = (bf*)alloc((size_t)B_ * T_ * D_ * 2);
  bf* w1b    = (bf*)alloc((size_t)MH_ * D_ * 2);
  bf* w2b    = (bf*)alloc((size_t)D_ * MH_ * 2);
  bf* wqkvb  = (bf*)alloc((size_t)3 * D_ * D_ * 2);
  bf* wob    = (bf*)alloc((size_t)D_ * D_ * 2);
  bf* fw1b   = (bf*)alloc((size_t)FF_ * D_ * 2);
  bf* fw2b   = (bf*)alloc((size_t)D_ * FF_ * 2);
  bf* memh   = (bf*)alloc((size_t)B_ * T_ * MH_ * 2);
  float* mctx= (float*)alloc((size_t)B_ * T_ * D_ * 4);
  float* aug = (float*)alloc((size_t)ROWS_ * D_ * 4);
  bf* hln    = (bf*)alloc((size_t)ROWS_ * D_ * 2);
  bf* qkvb   = (bf*)alloc((size_t)(ROWS_ + QKV_PAD_) * 3 * D_ * 2);
  bf* ctxb   = (bf*)alloc((size_t)ROWS_ * D_ * 2);
  float* hres= (float*)alloc((size_t)ROWS_ * D_ * 4);
  bf* fln    = (bf*)alloc((size_t)ROWS_ * D_ * 2);
  bf* gact   = (bf*)alloc((size_t)ROWS_ * FF_ * 2);
  (void)ws_size; (void)in_sizes; (void)n_in; (void)out_size;

  auto cvt = [&](const float* src, bf* dst, long n) {
    cvt_bf16_kernel<<<(unsigned)((n / 4 + 255) / 256), 256, 0, stream>>>(src, dst, n);
  };
  cvt(x,    xb,    (long)B_ * T_ * D_);
  cvt(mW1,  w1b,   (long)MH_ * D_);
  cvt(mW2,  w2b,   (long)D_ * MH_);
  cvt(Wqkv, wqkvb, (long)3 * D_ * D_);
  cvt(Wo,   wob,   (long)D_ * D_);
  cvt(fW1,  fw1b,  (long)FF_ * D_);
  cvt(fW2,  fw2b,  (long)D_ * FF_);

  // zero qkv padding rows (keeps attention clamp-free and WMMA NaN-safe)
  {
    long npad = (long)QKV_PAD_ * 3 * D_;
    zero_bf16_kernel<<<(unsigned)((npad / 8 + 255) / 256), 256, 0, stream>>>(
        qkvb + (size_t)ROWS_ * 3 * D_, npad);
  }

  dim3 blk(256);
  // 1. mem_h = relu(x @ mW1^T + mb1)          (2048 x 4096, K=1024)
  gemm_kernel<EPI_RELU_BF16><<<dim3(8, MH_ / 64), blk, 0, stream>>>(
      xb, w1b, mb1, nullptr, memh, B_ * T_, MH_, D_);
  // 2. mem_ctx = mem_h @ mW2^T + mb2          (2048 x 1024, K=4096)
  gemm_kernel<EPI_BIAS_F32><<<dim3(8, D_ / 64), blk, 0, stream>>>(
      memh, w2b, mb2, nullptr, mctx, B_ * T_, D_, MH_);
  // 3. aug = concat(pmem, mem_ctx, x)
  build_aug_kernel<<<ROWS_, 256, 0, stream>>>(x, pmem, mctx, aug);
  // 4. h = LN1(aug) -> bf16
  ln_kernel<<<ROWS_, 256, 0, stream>>>(aug, ln1g, ln1b, hln);
  // 5. qkv = h @ Wqkv^T + bqkv                (4104 x 3072, K=1024)
  gemm_kernel<EPI_BIAS_BF16><<<dim3((ROWS_ + 255) / 256, 3 * D_ / 64), blk, 0, stream>>>(
      hln, wqkvb, bqkv, nullptr, qkvb, ROWS_, 3 * D_, D_);
  // 6. sliding-window attention -> ctx (bf16)
  attn_kernel<<<dim3((L_ + 15) / 16, H_, B_), 32, 0, stream>>>(qkvb, ctxb);
  // 7. h_res = aug + ctx @ Wo^T + bo          (4104 x 1024, K=1024)
  gemm_kernel<EPI_RES_F32><<<dim3((ROWS_ + 255) / 256, D_ / 64), blk, 0, stream>>>(
      ctxb, wob, bo, aug, hres, ROWS_, D_, D_);
  // 8. f = LN2(h_res) -> bf16
  ln_kernel<<<ROWS_, 256, 0, stream>>>(hres, ln2g, ln2b, fln);
  // 9. g = gelu(f @ fW1^T + fb1)              (4104 x 4096, K=1024)
  gemm_kernel<EPI_GELU_BF16><<<dim3((ROWS_ + 255) / 256, FF_ / 64), blk, 0, stream>>>(
      fln, fw1b, fb1, nullptr, gact, ROWS_, FF_, D_);
  // 10. out = h_res + g @ fW2^T + fb2         (4104 x 1024, K=4096)
  gemm_kernel<EPI_RES_F32><<<dim3((ROWS_ + 255) / 256, D_ / 64), blk, 0, stream>>>(
      gact, fw2b, fb2, hres, out, ROWS_, D_, FF_);
}
